// message_passing_gnn_induct_57363583205559
// MI455X (gfx1250) — matrix operand order repaired and checked by reading, not compiled, not run on hardware
//
#include <hip/hip_runtime.h>
#include <hip/hip_bf16.h>

typedef __bf16 bf16;
typedef __attribute__((ext_vector_type(16))) __bf16 v16bf;
typedef __attribute__((ext_vector_type(2)))  __bf16 v2bf;
typedef __attribute__((ext_vector_type(8)))  float  v8f;

#define BM 64
#define BN 128
#define BK 32
#define LDST 40       // padded LDS row stride (elements) to break bank conflicts
#define SEG_TILES 8   // concat segment width in K-tiles (256/32)

__device__ __forceinline__ unsigned pk2bf(float a, float b) {
    v2bf v;
    v.x = (bf16)a;
    v.y = (bf16)b;
    return __builtin_bit_cast(unsigned, v);   // -> v_cvt_pk_bf16_f32
}

// ---------------------------------------------------------------------------
// Weight transpose + f32->bf16 convert:  W[K][N] row-major -> Wt[N][Kpad] bf16
// ---------------------------------------------------------------------------
__global__ void wtrans_kernel(const float* __restrict__ W, bf16* __restrict__ Wt,
                              int K, int N, int Kpad) {
    int idx = blockIdx.x * 256 + threadIdx.x;
    if (idx >= N * Kpad) return;
    int n = idx / Kpad;
    int k = idx - n * Kpad;
    float v = (k < K) ? W[(size_t)k * N + n] : 0.0f;
    Wt[idx] = (bf16)v;
}

__global__ void zero_kernel(float* __restrict__ p, long long n) {
    long long i = (long long)blockIdx.x * blockDim.x + threadIdx.x;
    long long stride = (long long)gridDim.x * blockDim.x;
    for (; i < n; i += stride) p[i] = 0.0f;
}

// ---------------------------------------------------------------------------
// bf16-WMMA GEMM, double-buffered LDS pipeline.
//   C[M][ldc] = epilogue( A[M][Ktot] @ Wt^T ),  Wt stored [N][Kpad] bf16.
// Block = 128 threads = 4 waves, block tile 64x128; each wave computes a
// 32x64 sub-tile = 2x4 v_wmma_f32_16x16x32_bf16 accumulators.
// A (f32) is staged through registers and converted with v_cvt_pk_bf16_f32;
// B (bf16 weights) is copied by CDNA5 async DMA: global_load_async_to_lds_b128
// (ASYNCcnt) -- zero VGPR footprint, synchronized by s_wait_asynccnt before
// the tile barrier. CONCAT=true: A = [A0|A1|A2], segment pointer switched by
// a uniform branch every SEG_TILES tiles. Out-of-range M rows are
// load-clamped (garbage rows never stored). The only partial final K-tile
// (encoder) is staged once, guarded, outside the loop. Requires KT >= 2.
// ---------------------------------------------------------------------------
template <bool CONCAT>
__global__ __launch_bounds__(128, 1) void gemm_bf16_kernel(
    const float* __restrict__ A0, const float* __restrict__ A1,
    const float* __restrict__ A2,
    int lda, int M, int Ktot, int KT /* K tiles of 32 */,
    const bf16* __restrict__ Wt,   // [Ntotal][Kpad], Kpad = KT*32
    const float* __restrict__ bias,
    const float* resid,            // nullable [M][ldc]
    float* C, int ldc, int relu)
{
    __shared__ bf16 As[2][BM * LDST];
    __shared__ bf16 Bs[2][BN * LDST];

    const int tid  = threadIdx.x;          // 0..127
    const int lane = tid & 31;
    const int wid  = tid >> 5;             // 0..3
    const int wm   = wid & 1;              // M strip (32 rows each)
    const int wn   = wid >> 1;             // N strip (64 cols each)
    const int m0   = blockIdx.y * BM;
    const int n0   = blockIdx.x * BN;
    const int Kpad = KT * BK;

    v8f acc[2][4];
    v8f vzero = {};
#pragma unroll
    for (int i = 0; i < 2; ++i)
#pragma unroll
        for (int j = 0; j < 4; ++j) acc[i][j] = vzero;

    // A staging: each thread moves 16 consecutive K-elements of one row
    const int srow = tid >> 1;             // 0..63
    const int scol = (tid & 1) * 16;       // 0 or 16
    const int grow = m0 + srow;
    const int growc = (grow < M) ? grow : (M - 1);  // clamp: no guard in loop

    const int r  = lane & 15;              // row/col within a 16x16 tile
    const int hi = lane >> 4;              // lane half selects K sub-block

    // pointer-increment addressing (no per-iteration 64-bit address math)
    const size_t rowOff = (size_t)growc * lda;
    const float2* aPtr = (const float2*)(A0 + rowOff + scol);
    int segIdx = 0;
    // B async DMA: each thread copies one full 32-wide K-chunk of one Wt row
    const bf16* bPtr = Wt + (size_t)(n0 + tid) * Kpad;
    const unsigned ldsB0 = (unsigned)(size_t)&Bs[0][tid * LDST];
    const unsigned ldsB1 = (unsigned)(size_t)&Bs[1][tid * LDST];

    float2 aF[8];   // raw staged A (convert at commit time)

    auto stageA = [&]() {
#pragma unroll
        for (int e = 0; e < 8; ++e) aF[e] = aPtr[e];
    };
    auto advanceA = [&](int justStaged) {
        if (CONCAT && ((justStaged + 1) % SEG_TILES == 0)) {   // uniform branch
            ++segIdx;
            const float* An = (segIdx == 1) ? A1 : A2;
            aPtr = (const float2*)(An + rowOff + scol);
        } else {
            aPtr += BK / 2;
        }
    };
    auto stageBasync = [&](int buf) {      // CDNA5 async DMA, no VGPR staging
        unsigned ldsOff = buf ? ldsB1 : ldsB0;
        unsigned long long ga = (unsigned long long)(size_t)bPtr;
        asm volatile(
            "global_load_async_to_lds_b128 %0, %1, off\n\t"
            "global_load_async_to_lds_b128 %0, %1, off offset:16\n\t"
            "global_load_async_to_lds_b128 %0, %1, off offset:32\n\t"
            "global_load_async_to_lds_b128 %0, %1, off offset:48"
            :: "v"(ldsOff), "v"(ga) : "memory");
        bPtr += BK;
    };
    auto waitAsync = [&]() {
        asm volatile("s_wait_asynccnt 0" ::: "memory");
    };
    auto commitA = [&](int buf) {          // after compute: wait lands here
        uint4 qa0, qa1;
        qa0.x = pk2bf(aF[0].x, aF[0].y);
        qa0.y = pk2bf(aF[1].x, aF[1].y);
        qa0.z = pk2bf(aF[2].x, aF[2].y);
        qa0.w = pk2bf(aF[3].x, aF[3].y);
        qa1.x = pk2bf(aF[4].x, aF[4].y);
        qa1.y = pk2bf(aF[5].x, aF[5].y);
        qa1.z = pk2bf(aF[6].x, aF[6].y);
        qa1.w = pk2bf(aF[7].x, aF[7].y);
        uint4* da = (uint4*)&As[buf][srow * LDST + scol];
        da[0] = qa0;
        da[1] = qa1;
    };
    auto stageAtail = [&](int kt) {        // encoder-only partial final tile
        const int kc = kt * BK + scol;
        float t[16];
#pragma unroll
        for (int e = 0; e < 16; ++e) {
            int k = kc + e;
            t[e] = (k < Ktot) ? A0[rowOff + k] : 0.0f;
        }
#pragma unroll
        for (int e = 0; e < 8; ++e) { aF[e].x = t[2 * e]; aF[e].y = t[2 * e + 1]; }
    };
    auto compute = [&](int buf) {
        // A fragments (reused 4x) per 16-bit WMMA A layout:
        // lane half hi holds K = hi*8..+7 then 16+hi*8..+7
        v16bf afrag[2];
#pragma unroll
        for (int i = 0; i < 2; ++i) {
            int row = wm * 32 + i * 16 + r;
            const bf16* p = &As[buf][row * LDST + hi * 8];
            union { v16bf v; uint4 q[2]; } u;
            u.q[0] = *(const uint4*)p;
            u.q[1] = *(const uint4*)(p + 16);
            afrag[i] = u.v;
        }
        // B 32x16 layout: lane half hi holds K = hi*8..+7 then 16+hi*8..+7;
        // one B fragment live at a time keeps peak VGPR pressure low
#pragma unroll
        for (int j = 0; j < 4; ++j) {
            int col = wn * 64 + j * 16 + r;
            const bf16* p = &Bs[buf][col * LDST + hi * 8];
            union { v16bf v; uint4 q[2]; } u;
            u.q[0] = *(const uint4*)p;
            u.q[1] = *(const uint4*)(p + 16);
            acc[0][j] = __builtin_amdgcn_wmma_f32_16x16x32_bf16(
                false, afrag[0], false, u.v, (short)0, acc[0][j], false, false);
            acc[1][j] = __builtin_amdgcn_wmma_f32_16x16x32_bf16(
                false, afrag[1], false, u.v, (short)0, acc[1][j], false, false);
        }
    };

    // prologue: tile 0 (always full since KT >= 2)
    stageA();
    advanceA(0);
    stageBasync(0);
    commitA(0);
    waitAsync();
    __syncthreads();

    // steady state over guaranteed-full tiles: branch-free body
    for (int kt = 0; kt < KT - 2; ++kt) {
        const int buf = kt & 1;
        stageBasync(buf ^ 1);      // async DMA into the other buffer
        stageA();                  // 8 loads, in flight through the WMMAs
        advanceA(kt + 1);
        compute(buf);
        commitA(buf ^ 1);          // waits for A loads after WMMA issue
        waitAsync();               // B DMA visible before the barrier
        __syncthreads();
    }
    // peeled iteration kt = KT-2: final tile may be partial (encoder only)
    {
        const int buf = (KT - 2) & 1;
        stageBasync(buf ^ 1);
        if (Ktot == Kpad) stageA();       // uniform branch, once
        else              stageAtail(KT - 1);
        compute(buf);
        commitA(buf ^ 1);
        waitAsync();
        __syncthreads();
    }
    compute((KT - 1) & 1);

    // epilogue: C/D layout = VGPR v -> M = hi*8+v, lane r -> N
#pragma unroll
    for (int i = 0; i < 2; ++i) {
#pragma unroll
        for (int j = 0; j < 4; ++j) {
            int col = n0 + wn * 64 + j * 16 + r;
            float b = bias[col];
#pragma unroll
            for (int v = 0; v < 8; ++v) {
                int row = m0 + wm * 32 + i * 16 + hi * 8 + v;
                if (row < M) {
                    float val = acc[i][j][v] + b;
                    if (relu) val = fmaxf(val, 0.0f);
                    if (resid) val += resid[(size_t)row * ldc + col];
                    C[(size_t)row * ldc + col] = val;
                }
            }
        }
    }
}

// ---------------------------------------------------------------------------
// Edge scatter: one wave per edge, both directions fused.
//   fi[dst] += xp[src];  fo[src] += xc[dst]
// ---------------------------------------------------------------------------
__global__ __launch_bounds__(256) void scatter_kernel(
    const float* __restrict__ xp, const float* __restrict__ xc,
    const int* __restrict__ src, const int* __restrict__ dst,
    float* fi, float* fo, int nE)
{
    int e = blockIdx.x * 8 + (threadIdx.x >> 5);
    if (e >= nE) return;
    int lane = threadIdx.x & 31;
    int s = src[e], d = dst[e];
    const float* ps = xp + (size_t)s * 256;
    const float* pd = xc + (size_t)d * 256;
    float* qd = fi + (size_t)d * 256;
    float* qs = fo + (size_t)s * 256;
#pragma unroll
    for (int f = lane; f < 256; f += 32) {
        atomicAdd(qd + f, ps[f]);
        atomicAdd(qs + f, pd[f]);
    }
}

// ---------------------------------------------------------------------------
extern "C" void kernel_launch(void* const* d_in, const int* in_sizes, int n_in,
                              void* d_out, int out_size, void* d_ws, size_t ws_size,
                              hipStream_t stream) {
    const int IN_DIM = 1000, D = 256;
    const int M = in_sizes[0] / IN_DIM;      // 100000 nodes
    const int E = in_sizes[1] / 2;           // 200000 edges

    const float* nodes  = (const float*)d_in[0];
    const int*   edges  = (const int*)d_in[1];   // [2][E]
    const float* W_enc  = (const float*)d_in[3];
    const float* b_enc  = (const float*)d_in[4];
    const float* W_p    = (const float*)d_in[5];
    const float* b_p    = (const float*)d_in[6];
    const float* W_c    = (const float*)d_in[7];
    const float* b_c    = (const float*)d_in[8];
    const float* W_f1   = (const float*)d_in[9];
    const float* b_f1   = (const float*)d_in[10];
    const float* W_f2   = (const float*)d_in[11];
    const float* b_f2   = (const float*)d_in[12];
    const float* W_conv = (const float*)d_in[13];
    const float* b_conv = (const float*)d_in[14];
    const int* src = edges;
    const int* dst = edges + E;

    // ---- workspace carve-up (256B aligned) ----
    size_t off = 0;
    char* base = (char*)d_ws;
    auto carve = [&](size_t bytes) -> void* {
        void* p = base + off;
        off += (bytes + 255) & ~(size_t)255;
        return p;
    };
    float* x   = (float*)carve((size_t)M * D * 4);
    float* xp  = (float*)carve((size_t)M * D * 4);
    float* xc  = (float*)carve((size_t)M * D * 4);
    float* fi  = (float*)carve((size_t)M * D * 4);   // fi,fo contiguous
    float* fo  = (float*)carve((size_t)M * D * 4);
    float* hid = (float*)carve((size_t)M * 2 * D * 4);
    bf16* WtE  = (bf16*)carve((size_t)256 * 1024 * 2);  // [256][1024]
    bf16* WtP  = (bf16*)carve((size_t)256 * 256 * 2);
    bf16* WtC  = (bf16*)carve((size_t)256 * 256 * 2);
    bf16* WtF1 = (bf16*)carve((size_t)512 * 768 * 2);
    bf16* WtF2 = (bf16*)carve((size_t)256 * 512 * 2);
    bf16* WtCv = (bf16*)carve((size_t)512 * 256 * 2);
    (void)ws_size; (void)n_in;

    // ---- weight transpose/convert to bf16 [N][Kpad] ----
    auto wt = [&](const float* W, bf16* Wt, int K, int N, int Kpad) {
        int tot = N * Kpad;
        wtrans_kernel<<<(tot + 255) / 256, 256, 0, stream>>>(W, Wt, K, N, Kpad);
    };
    wt(W_enc,  WtE,  IN_DIM, D,     1024);
    wt(W_p,    WtP,  D,      D,     D);
    wt(W_c,    WtC,  D,      D,     D);
    wt(W_f1,   WtF1, 3 * D,  2 * D, 3 * D);
    wt(W_f2,   WtF2, 2 * D,  D,     2 * D);
    wt(W_conv, WtCv, D,      2 * D, D);

    const int mTiles = (M + BM - 1) / BM;
    dim3 blk(128);

    // ---- encoder: x = nodes @ W_enc + b_enc ----
    gemm_bf16_kernel<false><<<dim3(D / BN, mTiles), blk, 0, stream>>>(
        nodes, nodes, nodes, /*lda=*/IN_DIM, M,
        /*Ktot=*/IN_DIM, /*KT=*/1024 / BK, WtE, b_enc,
        /*resid=*/nullptr, x, /*ldc=*/D, /*relu=*/0);

    // ---- 4 message-passing iterations ----
    for (int it = 0; it < 4; ++it) {
        // zero fi & fo (contiguous)
        zero_kernel<<<2048, 256, 0, stream>>>(fi, (long long)M * D * 2);

        // xp = relu(x @ W_p + b_p)   (per-node; gather moved into scatter)
        gemm_bf16_kernel<false><<<dim3(D / BN, mTiles), blk, 0, stream>>>(
            x, x, x, D, M, D, D / BK, WtP, b_p, nullptr, xp, D, 1);
        // xc = relu(x @ W_c + b_c)
        gemm_bf16_kernel<false><<<dim3(D / BN, mTiles), blk, 0, stream>>>(
            x, x, x, D, M, D, D / BK, WtC, b_c, nullptr, xc, D, 1);

        // fi[dst] += xp[src];  fo[src] += xc[dst]
        scatter_kernel<<<(E + 7) / 8, 256, 0, stream>>>(xp, xc, src, dst, fi, fo, E);

        // hid = relu([x|fi|fo] @ W_f1 + b_f1)   (concat fused via segments)
        gemm_bf16_kernel<true><<<dim3(2 * D / BN, mTiles), blk, 0, stream>>>(
            x, fi, fo, D, M, 3 * D, 3 * D / BK, WtF1, b_f1,
            nullptr, hid, 2 * D, 1);

        // x = x + (hid @ W_f2 + b_f2)   (residual fused)
        gemm_bf16_kernel<false><<<dim3(D / BN, mTiles), blk, 0, stream>>>(
            hid, hid, hid, 2 * D, M, 2 * D, 2 * D / BK, WtF2, b_f2,
            x, x, D, 0);
    }

    // ---- out = x @ W_conv + b_conv ----
    gemm_bf16_kernel<false><<<dim3(2 * D / BN, mTiles), blk, 0, stream>>>(
        x, x, x, D, M, D, D / BK, WtCv, b_conv,
        nullptr, (float*)d_out, 2 * D, 0);
}
